// LSTMDecoder_17480516895252
// MI455X (gfx1250) — compile-verified, matrix operand order checked by reading, not run on hardware
//
#include <hip/hip_runtime.h>
#include <math.h>

// Problem dims
#define B_ 64
#define S_ 1024
#define H_ 512
#define E_ 512
#define V_ 32000
#define P_ 40

typedef __attribute__((ext_vector_type(16))) __bf16 v16bf;
typedef __attribute__((ext_vector_type(8)))  __bf16 bf16x8;
typedef __attribute__((ext_vector_type(8)))  float  v8f;

union Frag16 { v16bf v; bf16x8 h[2]; };

static __device__ __forceinline__ __bf16 f2bf(float f) {
  unsigned u = __float_as_uint(f);
  u += 0x7FFFu + ((u >> 16) & 1u);            // round-to-nearest-even
  unsigned short hs = (unsigned short)(u >> 16);
  return __builtin_bit_cast(__bf16, hs);
}
static __device__ __forceinline__ float sigf(float x) { return 1.0f / (1.0f + __expf(-x)); }

// gfx1250 has a hardware v_tanh_f32 (TRANS op); fall back to exp-based tanh.
static __device__ __forceinline__ float tanh_fast(float x) {
#if __has_builtin(__builtin_amdgcn_tanhf)
  return __builtin_amdgcn_tanhf(x);
#else
  float e = __expf(2.0f * x);
  return 1.0f - 2.0f / (e + 1.0f);
#endif
}

// ---------------------------------------------------------------- W_h -> bf16
__global__ void k_cvt_wh(const float* __restrict__ wh, __bf16* __restrict__ whB) {
  int i = blockIdx.x * 256 + threadIdx.x;     // 512*512 elems
  whB[i] = f2bf(wh[i]);
}

// ------------------------------------------------- qs = hidden @ W_s.T (B,H)
__global__ void k_qs(const float* __restrict__ hid, const float* __restrict__ Ws,
                     float* __restrict__ qs) {
  int idx = blockIdx.x * 256 + threadIdx.x;   // B*H threads
  int b = idx >> 9, k = idx & (H_ - 1);
  const float* hrow = hid + b * H_;
  const float* wrow = Ws + (size_t)k * H_;
  float acc = 0.f;
  for (int h = 0; h < H_; ++h) acc += hrow[h] * wrow[h];
  qs[idx] = acc;
}

// ------------- fused attention-energy GEMM: scores[b,s] = Vw . tanh(qs + eh)
// grid: (B*S/32) blocks, 256 threads (8 waves). One block = 32 rows x 512 cols.
__global__ void __launch_bounds__(256)
k_attn_scores(const float* __restrict__ enc, const __bf16* __restrict__ whB,
              const float* __restrict__ qs, const float* __restrict__ Vw,
              float* __restrict__ scores) {
  // bf16 staging tile and f32 eh tile share the same LDS (used in disjoint phases)
  // alignas(16) so bf16x8 (ds_load_b128) accesses keep provable 16B alignment
  __shared__ alignas(16) union {
    __bf16 eT[32][H_ + 8];     // 32 x 520 bf16 (33,280 B), row stride 1040 B (16B mult)
    float  ehF[32][H_ + 1];    // 32 x 513 f32 (65,664 B), stride%banks==1
  } sm;
  __shared__ float qsL[H_];
  __shared__ float vwL[H_];
  __shared__ float red[256];

  const int tid = threadIdx.x;
  const int b = blockIdx.x >> 5;              // 32 tiles per batch row
  const int s0 = (blockIdx.x & 31) * 32;

  // stage 32 encoder rows (f32 -> bf16) + qs row + Vw
  const float* src = enc + ((size_t)b * S_ + s0) * H_;
  for (int i = 0; i < 64; ++i) {
    int idx = tid + i * 256;                  // 32*512 elems
    int r = idx >> 9, c = idx & (H_ - 1);
    sm.eT[r][c] = f2bf(src[(size_t)r * H_ + c]);
  }
  qsL[tid] = qs[b * H_ + tid];
  qsL[tid + 256] = qs[b * H_ + tid + 256];
  vwL[tid] = Vw[tid];
  vwL[tid + 256] = Vw[tid + 256];
  __syncthreads();

  const int wave = tid >> 5, lane = tid & 31;
  const int lm = lane & 15;
  const int ksel = (lane >> 4) & 1;           // K sub-chunk select per ISA layout
  const int mofs = (lane >= 16) ? 8 : 0;

  // this wave owns output columns [wave*64, wave*64+64): 4 N-tiles
  const __bf16* wrow[4];
  int ncol[4];
  for (int j = 0; j < 4; ++j) {
    ncol[j] = wave * 64 + j * 16 + lm;
    wrow[j] = whB + (size_t)ncol[j] * H_;
  }

  v8f acc[2][4];
  for (int mt = 0; mt < 2; ++mt)
    for (int j = 0; j < 4; ++j) acc[mt][j] = (v8f)(0.f);

  for (int kk = 0; kk < 16; ++kk) {
    const int k0 = kk * 32;
    Frag16 a0, a1;
    a0.h[0] = *(const bf16x8*)&sm.eT[lm][k0 + ksel * 8];
    a0.h[1] = *(const bf16x8*)&sm.eT[lm][k0 + 16 + ksel * 8];
    a1.h[0] = *(const bf16x8*)&sm.eT[16 + lm][k0 + ksel * 8];
    a1.h[1] = *(const bf16x8*)&sm.eT[16 + lm][k0 + 16 + ksel * 8];
    for (int j = 0; j < 4; ++j) {
      Frag16 bf;
      bf.h[0] = *(const bf16x8*)(wrow[j] + k0 + ksel * 8);
      bf.h[1] = *(const bf16x8*)(wrow[j] + k0 + 16 + ksel * 8);
      acc[0][j] = __builtin_amdgcn_wmma_f32_16x16x32_bf16(
          false, a0.v, false, bf.v, (short)0, acc[0][j], false, false);
      acc[1][j] = __builtin_amdgcn_wmma_f32_16x16x32_bf16(
          false, a1.v, false, bf.v, (short)0, acc[1][j], false, false);
    }
  }
  __syncthreads();   // all waves done reading eT; reuse LDS for f32 eh tile

  // dump accumulators to LDS (consumes them immediately -> no scratch spill)
  for (int j = 0; j < 4; ++j)
    for (int mt = 0; mt < 2; ++mt)
      for (int r = 0; r < 8; ++r)
        sm.ehF[mt * 16 + mofs + r][ncol[j]] = acc[mt][j][r];
  __syncthreads();

  // reduction pass: 8 threads per row, 64 columns each
  const int row = tid >> 3;
  const int c0 = (tid & 7) * 64;
  float p = 0.f;
  for (int c = c0; c < c0 + 64; ++c)
    p += tanh_fast(sm.ehF[row][c] + qsL[c]) * vwL[c];
  red[tid] = p;
  __syncthreads();
  if ((tid & 7) == 0) {
    float s = red[tid] + red[tid + 1] + red[tid + 2] + red[tid + 3] +
              red[tid + 4] + red[tid + 5] + red[tid + 6] + red[tid + 7];
    scores[(size_t)b * S_ + s0 + row] = s;
  }
}

// --------------------------- POS gate + masked softmax over S; writes attn out
__global__ void __launch_bounds__(256)
k_softmax_gate(const float* __restrict__ scores, const float* __restrict__ pos_oh,
               const unsigned char* __restrict__ mask, const float* __restrict__ Wp,
               const float* __restrict__ bp, float* __restrict__ attn) {
  __shared__ float sv[S_];
  __shared__ float red[256];
  const int b = blockIdx.x, tid = threadIdx.x;
  const float bias = bp[0];
  for (int s = tid; s < S_; s += 256) {
    const float* po = pos_oh + ((size_t)b * S_ + s) * P_;
    float g = bias;
    for (int p = 0; p < P_; ++p) g += po[p] * Wp[p];
    float sc = scores[(size_t)b * S_ + s] * sigf(g);
    if (mask[(size_t)b * S_ + s]) sc = -1e30f;
    sv[s] = sc;
  }
  __syncthreads();
  float m = -3.4e38f;
  for (int s = tid; s < S_; s += 256) m = fmaxf(m, sv[s]);
  red[tid] = m;
  __syncthreads();
  for (int o = 128; o > 0; o >>= 1) {
    if (tid < o) red[tid] = fmaxf(red[tid], red[tid + o]);
    __syncthreads();
  }
  m = red[0];
  __syncthreads();
  float part = 0.f;
  for (int s = tid; s < S_; s += 256) { float e = __expf(sv[s] - m); sv[s] = e; part += e; }
  red[tid] = part;
  __syncthreads();
  for (int o = 128; o > 0; o >>= 1) {
    if (tid < o) red[tid] += red[tid + o];
    __syncthreads();
  }
  const float inv = 1.f / red[0];
  for (int s = tid; s < S_; s += 256) attn[(size_t)b * S_ + s] = sv[s] * inv;
}

// ------------------------------------- context[b,h] = sum_s attn[b,s]*enc[b,s,h]
__global__ void __launch_bounds__(256)
k_context(const float* __restrict__ attn, const float* __restrict__ enc,
          float* __restrict__ ctx) {
  __shared__ float aL[S_];
  const int b = blockIdx.x, tid = threadIdx.x;
  for (int s = tid; s < S_; s += 256) aL[s] = attn[(size_t)b * S_ + s];
  __syncthreads();
  const float* e = enc + (size_t)b * S_ * H_;
  float a0 = 0.f, a1 = 0.f;
  for (int s = 0; s < S_; ++s) {
    float w = aL[s];
    a0 += w * e[(size_t)s * H_ + tid];
    a1 += w * e[(size_t)s * H_ + tid + 256];
  }
  ctx[b * H_ + tid] = a0;
  ctx[b * H_ + tid + 256] = a1;
}

// ------------- LSTM pre-activations: gates = [emb|ctx]@W_ih.T + h@W_hh.T + b
__global__ void __launch_bounds__(256)
k_gates(const int* __restrict__ tok, const float* __restrict__ embT,
        const float* __restrict__ ctx, const float* __restrict__ hid,
        const float* __restrict__ Wih, const float* __restrict__ Whh,
        const float* __restrict__ bih, const float* __restrict__ bhh,
        float* __restrict__ gates) {
  const int idx = blockIdx.x * 256 + threadIdx.x;   // B*4H
  const int b = idx >> 11, j = idx & 2047;
  const float* wih = Wih + (size_t)j * (E_ + H_);
  const float* whh = Whh + (size_t)j * H_;
  const float* emb = embT + (size_t)tok[b] * E_;
  const float* cx = ctx + b * H_;
  const float* hd = hid + b * H_;
  float acc = bih[j] + bhh[j];
  for (int e = 0; e < E_; ++e) acc += emb[e] * wih[e];
  for (int h = 0; h < H_; ++h) acc += cx[h] * wih[E_ + h];
  for (int h = 0; h < H_; ++h) acc += hd[h] * whh[h];
  gates[idx] = acc;
}

// ---------------------------------------------------------- cell/hidden update
__global__ void k_cell(const float* __restrict__ gates, const float* __restrict__ cell,
                       float* __restrict__ h1, float* __restrict__ c1) {
  const int idx = blockIdx.x * 256 + threadIdx.x;   // B*H
  const int b = idx >> 9, h = idx & (H_ - 1);
  const float* g = gates + (size_t)b * 4 * H_;
  float gi = g[h], gf = g[H_ + h], gg = g[2 * H_ + h], go = g[3 * H_ + h];
  float c = sigf(gf) * cell[idx] + sigf(gi) * tanh_fast(gg);
  float hh = sigf(go) * tanh_fast(c);
  c1[idx] = c;
  h1[idx] = hh;
}

// --------------- pred = h1 @ W_out.T + b_out via WMMA; W_out read exactly once
// grid: V/128 = 250 blocks, 256 threads (8 waves). Block = 64 rows x 128 cols.
__global__ void __launch_bounds__(256)
k_pred(const float* __restrict__ h1, const float* __restrict__ Wout,
       const float* __restrict__ bout, float* __restrict__ pred) {
  __shared__ alignas(16) __bf16 hT[B_][H_ + 8];   // row stride 1040 B (16B mult)
  const int tid = threadIdx.x;
  for (int i = 0; i < 128; ++i) {                 // 64*512 elems
    int idx = tid + i * 256;
    int r = idx >> 9, c = idx & (H_ - 1);
    hT[r][c] = f2bf(h1[(size_t)r * H_ + c]);
  }
  __syncthreads();

  const int wave = tid >> 5, lane = tid & 31;
  const int lm = lane & 15;
  const int ksel = (lane >> 4) & 1;
  const int mofs = (lane >= 16) ? 8 : 0;
  const int nbase = blockIdx.x * 128 + wave * 16;
  const int ncol = nbase + lm;
  const float* wrow = Wout + (size_t)ncol * H_;

  v8f acc[4];
  for (int mt = 0; mt < 4; ++mt) acc[mt] = (v8f)(0.f);

  for (int kk = 0; kk < 16; ++kk) {
    const int k0 = kk * 32;
    Frag16 bf;
    const float4* p0 = (const float4*)(wrow + k0 + ksel * 8);
    const float4* p1 = (const float4*)(wrow + k0 + 16 + ksel * 8);
    float4 x0 = p0[0], x1 = p0[1], y0 = p1[0], y1 = p1[1];
    bf.v[0] = f2bf(x0.x);  bf.v[1] = f2bf(x0.y);  bf.v[2] = f2bf(x0.z);  bf.v[3] = f2bf(x0.w);
    bf.v[4] = f2bf(x1.x);  bf.v[5] = f2bf(x1.y);  bf.v[6] = f2bf(x1.z);  bf.v[7] = f2bf(x1.w);
    bf.v[8] = f2bf(y0.x);  bf.v[9] = f2bf(y0.y);  bf.v[10] = f2bf(y0.z); bf.v[11] = f2bf(y0.w);
    bf.v[12] = f2bf(y1.x); bf.v[13] = f2bf(y1.y); bf.v[14] = f2bf(y1.z); bf.v[15] = f2bf(y1.w);
    for (int mt = 0; mt < 4; ++mt) {
      Frag16 af;
      af.h[0] = *(const bf16x8*)&hT[mt * 16 + lm][k0 + ksel * 8];
      af.h[1] = *(const bf16x8*)&hT[mt * 16 + lm][k0 + 16 + ksel * 8];
      acc[mt] = __builtin_amdgcn_wmma_f32_16x16x32_bf16(
          false, af.v, false, bf.v, (short)0, acc[mt], false, false);
    }
  }

  const float bo = bout[ncol];
  for (int mt = 0; mt < 4; ++mt)
    for (int r = 0; r < 8; ++r) {
      int row = mt * 16 + mofs + r;
      pred[(size_t)row * V_ + ncol] = acc[mt][r] + bo;
    }
}

// ---------------------------------------------------------------------- launch
extern "C" void kernel_launch(void* const* d_in, const int* in_sizes, int n_in,
                              void* d_out, int out_size, void* d_ws, size_t ws_size,
                              hipStream_t stream) {
  (void)in_sizes; (void)n_in; (void)out_size; (void)ws_size;
  const int*   tok    = (const int*)d_in[0];
  const float* hidden = (const float*)d_in[1];
  const float* cell   = (const float*)d_in[2];
  const float* enc    = (const float*)d_in[3];
  const float* pos_oh = (const float*)d_in[4];
  const unsigned char* mask = (const unsigned char*)d_in[5];
  const float* embT   = (const float*)d_in[6];
  const float* Ws     = (const float*)d_in[7];
  const float* Wh     = (const float*)d_in[8];
  const float* Vw     = (const float*)d_in[9];
  const float* Wp     = (const float*)d_in[10];
  const float* bp     = (const float*)d_in[11];
  const float* Wih    = (const float*)d_in[12];
  const float* Whh    = (const float*)d_in[13];
  const float* bih    = (const float*)d_in[14];
  const float* bhh    = (const float*)d_in[15];
  const float* Wout   = (const float*)d_in[16];
  const float* bout   = (const float*)d_in[17];

  float* out  = (float*)d_out;
  float* pred = out;                       // B*V
  float* h1   = out + (size_t)B_ * V_;     // B*H
  float* c1   = h1 + B_ * H_;              // B*H
  float* attn = c1 + B_ * H_;              // B*S

  char* ws = (char*)d_ws;
  float*  qs     = (float*)(ws);                       // B*H      (128 KB)
  float*  scores = (float*)(ws + 131072);              // B*S      (256 KB)
  float*  ctx    = (float*)(ws + 393216);              // B*H      (128 KB)
  float*  gates  = (float*)(ws + 524288);              // B*4H     (512 KB)
  __bf16* whB    = (__bf16*)(ws + 1048576);            // H*H bf16 (512 KB)

  k_cvt_wh<<<(H_ * H_) / 256, 256, 0, stream>>>(Wh, whB);
  k_qs<<<(B_ * H_) / 256, 256, 0, stream>>>(hidden, Ws, qs);
  k_attn_scores<<<(B_ * S_) / 32, 256, 0, stream>>>(enc, whB, qs, Vw, scores);
  k_softmax_gate<<<B_, 256, 0, stream>>>(scores, pos_oh, mask, Wp, bp, attn);
  k_context<<<B_, 256, 0, stream>>>(attn, enc, ctx);
  k_gates<<<(B_ * 4 * H_) / 256, 256, 0, stream>>>(tok, embT, ctx, hidden, Wih, Whh,
                                                   bih, bhh, gates);
  k_cell<<<(B_ * H_) / 256, 256, 0, stream>>>(gates, cell, h1, c1);
  k_pred<<<V_ / 128, 256, 0, stream>>>(h1, Wout, bout, pred);
}
